// RSSM_81810537054223
// MI455X (gfx1250) — matrix operand order repaired and checked by reading, not compiled, not run on hardware
//
#include <hip/hip_runtime.h>
#include <math.h>
#include <stdint.h>

// ---------------- problem constants ----------------
#define TT   64
#define BB   512
#define OBS  1024
#define ACT  6
#define STOCH 128
#define HID  1024
#define DH   512
#define KXA  160          // STOCH+ACT=134 padded to multiple of 32
#define LOG2PI 1.8378770664093453f

// ---------------- WMMA types ----------------
typedef __attribute__((ext_vector_type(16))) _Float16 v16h;
typedef __attribute__((ext_vector_type(8)))  float    v8f;

union Frag16 { v16h v; unsigned int u[8]; };

// ---------------- GEMM: C[M,N] = act([A1|A2][M,K] @ W[K,N] + bias) ----------------
// A sources are f16 (row strides % 8 == 0, d1 % 8 == 0), W pre-packed transposed
// Wt[N,K] f16.  M fixed 512 (grid.y = 4, tile 128).  K % 32 == 0.
// Staging: global_load_async_to_lds_b128, double-buffered, 3 async ops/wave/stage.
// N tails handled by clamping the B tile row (junk cols are never stored).
// mode: 0 = f16 store, 1 = f16 store + ELU, 2 = f32 store + fused NLL, 3 = f32 store.
#define GTM 128
#define GTN 64
#define GTK 32
#define GLDA 40   // padded f16 stride (80B/row)
#define GLDB 40
#define ASTRIDE (GTM * GLDA)
#define BSTRIDE (GTN * GLDB)

__device__ __forceinline__ void async_cp16(unsigned lds_off, const void* g) {
  asm volatile("global_load_async_to_lds_b128 %0, %1, off"
               :: "v"(lds_off), "v"((unsigned long long)(uintptr_t)g)
               : "memory");
}

__global__ __launch_bounds__(256) void gemm_wmma_kernel(
    const _Float16* __restrict__ A1, int lda1, int d1,
    const _Float16* __restrict__ A2, int lda2,
    const _Float16* __restrict__ Wt,
    const float* __restrict__ bias, void* __restrict__ Cout,
    int N, int K, int ldc, int mode,
    const float* __restrict__ tgt, float* __restrict__ acc_out)
{
  __shared__ __align__(16) _Float16 As[2 * ASTRIDE];
  __shared__ __align__(16) _Float16 Bs[2 * BSTRIDE];
  __shared__ float red[256];

  const int tid  = threadIdx.x;
  const int wave = tid >> 5;
  const int lane = tid & 31;
  const int half = lane >> 4;   // 0: lanes 0-15, 1: lanes 16-31
  const int r16  = lane & 15;
  const int m0 = blockIdx.y * GTM;
  const int n0 = blockIdx.x * GTN;

  // per-thread staging geometry
  // A tile 128x32 f16 = 512 chunks of 16B; 2 chunks per thread
  const int arow0 = (tid) >> 2,        aq0 = tid & 3;
  const int arow1 = (256 + tid) >> 2,  aq1 = tid & 3;   // rows 64..127
  // B tile 64x32 f16 = 256 chunks of 16B; 1 per thread, clamped row
  const int brow = tid >> 2, bq = tid & 3;
  int bn = n0 + brow; if (bn > N - 1) bn = N - 1;
  const _Float16* bgl = Wt + (size_t)bn * K + bq * 8;
  const unsigned aLds0 = (unsigned)(uintptr_t)&As[arow0 * GLDA + aq0 * 8];
  const unsigned aLds1 = (unsigned)(uintptr_t)&As[arow1 * GLDA + aq1 * 8];
  const unsigned bLds  = (unsigned)(uintptr_t)&Bs[brow * GLDB + bq * 8];

  auto stage = [&](int buf, int k0) {
    int c0 = k0 + aq0 * 8;       // aq0 == aq1
    const _Float16* g0 = (c0 < d1) ? &A1[(size_t)(m0 + arow0) * lda1 + c0]
                                   : &A2[(size_t)(m0 + arow0) * lda2 + (c0 - d1)];
    const _Float16* g1 = (c0 < d1) ? &A1[(size_t)(m0 + arow1) * lda1 + c0]
                                   : &A2[(size_t)(m0 + arow1) * lda2 + (c0 - d1)];
    unsigned boff = buf ? (unsigned)(ASTRIDE * sizeof(_Float16)) : 0u;
    async_cp16(aLds0 + boff, g0);
    async_cp16(aLds1 + boff, g1);
    unsigned bboff = buf ? (unsigned)(BSTRIDE * sizeof(_Float16)) : 0u;
    async_cp16(bLds + bboff, bgl + k0);
  };

  v8f acc[4];
#pragma unroll
  for (int i = 0; i < 4; ++i)
#pragma unroll
    for (int j = 0; j < 8; ++j) acc[i][j] = 0.0f;

  const int nstages = K / GTK;
  stage(0, 0);
  for (int st = 0; st < nstages; ++st) {
    const int cur = st & 1;
    const bool more = (st + 1 < nstages);
    if (more) stage(cur ^ 1, (st + 1) * GTK);
    if (more) asm volatile("s_wait_asynccnt 0x3" ::: "memory");
    else      asm volatile("s_wait_asynccnt 0x0" ::: "memory");
    __syncthreads();   // current tiles visible to all waves

    const _Float16* AsC = As + cur * ASTRIDE;
    const _Float16* BsC = Bs + cur * BSTRIDE;

    // A fragment (16x32 f16), ISA 7.12.2: lane m = r16,
    //   VGPR v holds K = (v>>2)*16 + half*8 + (v&3)*2 .. +1
    Frag16 af;
#pragma unroll
    for (int v = 0; v < 8; ++v) {
      int kk = ((v >> 2) << 4) + (half << 3) + ((v & 3) << 1);
      af.u[v] = *(const unsigned int*)&AsC[(wave * 16 + r16) * GLDA + kk];
    }
#pragma unroll
    for (int nt = 0; nt < 4; ++nt) {
      // B fragment (32x16 f16): same K order as A, VGPR v -> K = (v>>2)*16 + half*8 + (v&3)*2
      Frag16 bf;
#pragma unroll
      for (int v = 0; v < 8; ++v) {
        int kk = ((v >> 2) << 4) + (half << 3) + ((v & 3) << 1);
        bf.u[v] = *(const unsigned int*)&BsC[(nt * 16 + r16) * GLDB + kk];
      }
      acc[nt] = __builtin_amdgcn_wmma_f32_16x16x32_f16(
          false, af.v, false, bf.v, (short)0, acc[nt], false, false);
    }
    __syncthreads();   // all waves done reading tiles before next overwrite
  }

  // epilogue: C/D layout — lane 0-15: N = lane ; VGPR j: M = j / M = 8+j (lanes 16-31)
  float nll = 0.0f;
#pragma unroll
  for (int nt = 0; nt < 4; ++nt) {
    int nn = n0 + nt * 16 + r16;
    if (nn < N) {
      float bb = bias[nn];
#pragma unroll
      for (int j = 0; j < 8; ++j) {
        int mm = m0 + wave * 16 + half * 8 + j;
        float x = acc[nt][j] + bb;
        if (mode == 1) x = (x > 0.0f) ? x : (expf(x) - 1.0f);   // ELU
        if (mode <= 1) {
          ((_Float16*)Cout)[(size_t)mm * ldc + nn] = (_Float16)x;
        } else {
          ((float*)Cout)[(size_t)mm * ldc + nn] = x;
          if (mode == 2) {
            float d = tgt[(size_t)mm * ldc + nn] - x;
            nll += 0.5f * d * d;
          }
        }
      }
    }
  }
  if (mode == 2) {
    red[tid] = nll;
    __syncthreads();
    for (int o = 128; o > 0; o >>= 1) {
      if (tid < o) red[tid] += red[tid + o];
      __syncthreads();
    }
    if (tid == 0) atomicAdd(acc_out, red[0]);
  }
}

// ---------------- weight pack: W[K,N] f32 -> Wt[N,Kdst] f16 (Kdst >= K) ----------------
__global__ void pack_weight_kernel(const float* __restrict__ W, _Float16* __restrict__ Wt,
                                   int K, int N, int Kdst)
{
  int i = blockIdx.x * blockDim.x + threadIdx.x;
  if (i >= K * N) return;
  int k = i / N, n = i % N;
  Wt[(size_t)n * Kdst + k] = (_Float16)W[i];
}

// ---------------- f32 -> f16 bulk convert (float4 -> 4x f16, 8B stores) ----------------
__global__ void f32_to_f16_kernel(const float4* __restrict__ src, _Float16* __restrict__ dst,
                                  long n4)
{
  for (long i = blockIdx.x * (long)blockDim.x + threadIdx.x; i < n4;
       i += (long)gridDim.x * blockDim.x) {
    float4 v = src[i];
    union { _Float16 h[4]; unsigned long long u; } p;
    p.h[0] = (_Float16)v.x; p.h[1] = (_Float16)v.y;
    p.h[2] = (_Float16)v.z; p.h[3] = (_Float16)v.w;
    *(unsigned long long*)&dst[i * 4] = p.u;
  }
}

// ---------------- strided concat along feature dim ----------------
template <typename T>
__global__ void concat2_kernel(T* __restrict__ dst, int ldd,
                               const T* __restrict__ s1, int d1,
                               const T* __restrict__ s2, int d2, int rows)
{
  int d = d1 + d2;
  int n = rows * d;
  for (int i = blockIdx.x * blockDim.x + threadIdx.x; i < n; i += gridDim.x * blockDim.x) {
    int r = i / d, c = i % d;
    dst[(size_t)r * ldd + c] = (c < d1) ? s1[(size_t)r * d1 + c] : s2[(size_t)r * d2 + (c - d1)];
  }
}

// ---------------- GRU gate fusion (PyTorch r,z,n order), f16 state ----------------
__global__ void gru_update_kernel(_Float16* __restrict__ h,
                                  const _Float16* __restrict__ gi,
                                  const _Float16* __restrict__ gh)
{
  int i = blockIdx.x * blockDim.x + threadIdx.x;
  if (i >= BB * HID) return;
  int b = i >> 10, n = i & (HID - 1);
  const _Float16* gib = gi + (size_t)b * 3 * HID;
  const _Float16* ghb = gh + (size_t)b * 3 * HID;
  float ir = (float)gib[n], iz = (float)gib[HID + n], in = (float)gib[2 * HID + n];
  float hr = (float)ghb[n], hz = (float)ghb[HID + n], hn = (float)ghb[2 * HID + n];
  float rg = 1.0f / (1.0f + expf(-(ir + hr)));
  float zg = 1.0f / (1.0f + expf(-(iz + hz)));
  float ng = tanhf(in + rg * hn);
  h[i] = (_Float16)((1.0f - zg) * ng + zg * (float)h[i]);
}

__device__ __forceinline__ float softplusf(float x) {
  return (x > 20.0f) ? x : log1pf(expf(x));
}

// ---------------- posterior sample (f16 out) + KL reduce ----------------
__global__ void sample_kl_kernel(const float* __restrict__ pri, const float* __restrict__ pos,
                                 const float* __restrict__ eps, _Float16* __restrict__ s,
                                 float* __restrict__ acc)
{
  __shared__ float sm[256];
  int i = blockIdx.x * 256 + threadIdx.x;         // exactly BB*STOCH threads
  int b = i >> 7, j = i & (STOCH - 1);
  float mu_p = pri[(size_t)b * 2 * STOCH + j];
  float sp_r = pri[(size_t)b * 2 * STOCH + STOCH + j];
  float mu_q = pos[(size_t)b * 2 * STOCH + j];
  float sq_r = pos[(size_t)b * 2 * STOCH + STOCH + j];
  float sp = softplusf(sp_r) + 0.01f;
  float sq = softplusf(sq_r) + 0.01f;
  s[i] = (_Float16)(mu_q + sq * eps[i]);
  float d  = mu_q - mu_p;
  float kl = logf(sp / sq) + (sq * sq + d * d) / (2.0f * sp * sp) - 0.5f;
  sm[threadIdx.x] = kl;
  __syncthreads();
  for (int o = 128; o > 0; o >>= 1) {
    if (threadIdx.x < o) sm[threadIdx.x] += sm[threadIdx.x + o];
    __syncthreads();
  }
  if (threadIdx.x == 0) atomicAdd(acc, sm[0]);
}

__global__ void final_loss_kernel(const float* __restrict__ accs, float* __restrict__ out)
{
  const float denom = (float)(TT * BB);
  out[0] = (accs[0] + accs[1]) / denom + 0.5f * LOG2PI * (float)(OBS + ACT + 1);
}

static inline int ceil_div(int a, int b) { return (a + b - 1) / b; }

extern "C" void kernel_launch(void* const* d_in, const int* in_sizes, int n_in,
                              void* d_out, int out_size, void* d_ws, size_t ws_size,
                              hipStream_t stream) {
  (void)in_sizes; (void)n_in; (void)out_size; (void)ws_size;
  // ---- inputs, setup_inputs() dict (insertion) order ----
  const float* obs    = (const float*)d_in[0];   // T,B,OBS
  const float* action = (const float*)d_in[1];   // T,B,ACT
  const float* reward = (const float*)d_in[2];   // T,B,1
  const float* eps    = (const float*)d_in[3];   // T,B,STOCH
  const float* Wih  = (const float*)d_in[4];     // (134, 3072)
  const float* Whh  = (const float*)d_in[5];     // (1024,3072)
  const float* bih  = (const float*)d_in[6];
  const float* bhh  = (const float*)d_in[7];
  const float* pW[5][3]; const float* pB[5][3];
  for (int m = 0; m < 5; ++m)
    for (int l = 0; l < 3; ++l) {
      pW[m][l] = (const float*)d_in[8 + m * 6 + l * 2];
      pB[m][l] = (const float*)d_in[8 + m * 6 + l * 2 + 1];
    }
  // m: 0=post (2048->512->512->256), 1=prior (1024->..->256),
  //    2=obs_pre (1152->..->1024), 3=action_pre (1152->..->6), 4=reward_pre (1152->..->1)
  static const int mlpK[5][3] = {{2048,512,512},{1024,512,512},{1152,512,512},{1152,512,512},{1152,512,512}};
  static const int mlpN[5][3] = {{512,512,256}, {512,512,256}, {512,512,1024},{512,512,6},  {512,512,1}};

  // ---- workspace carve-out ----
  char* ws = (char*)d_ws;
  size_t cur = 0;
  auto alloc = [&](size_t bytes) -> char* {
    char* p = ws + cur;
    cur = (cur + bytes + 255) & ~(size_t)255;
    return p;
  };
  auto packTo = [&](_Float16* dst, const float* W, int K, int N, int Kdst) {
    pack_weight_kernel<<<ceil_div(K * N, 256), 256, 0, stream>>>(W, dst, K, N, Kdst);
  };
  auto pack = [&](const float* W, int K, int N) -> _Float16* {
    _Float16* dst = (_Float16*)alloc((size_t)K * N * sizeof(_Float16));
    packTo(dst, W, K, N, K);
    return dst;
  };

  // GRU input weights padded K=134 -> 160 (zero fill)
  _Float16* Wih_t = (_Float16*)alloc((size_t)KXA * 3 * HID * sizeof(_Float16));
  hipMemsetAsync(Wih_t, 0, (size_t)KXA * 3 * HID * sizeof(_Float16), stream);
  packTo(Wih_t, Wih, STOCH + ACT, 3 * HID, KXA);
  _Float16* Whh_t = pack(Whh, HID, 3 * HID);

  _Float16* mlpWt[5][3];
  for (int m = 0; m < 5; ++m)
    for (int l = 0; l < 3; ++l)
      mlpWt[m][l] = (l == 0 && (m == 2 || m == 4)) ? nullptr : pack(pW[m][l], mlpK[m][l], mlpN[m][l]);

  // fused decoder layer-0: rows [0,512) = obs_pre W0, rows [512,1024) = reward_pre W0
  _Float16* Wdec0 = (_Float16*)alloc((size_t)1024 * 1152 * sizeof(_Float16));
  packTo(Wdec0,                      pW[2][0], 1152, 512, 1152);
  packTo(Wdec0 + (size_t)512 * 1152, pW[4][0], 1152, 512, 1152);
  float* bdec0 = (float*)alloc(1024 * 4);
  concat2_kernel<float><<<4, 256, 0, stream>>>(bdec0, 1024, pB[2][0], 512, pB[4][0], 512, 1);

  // f16 copies of obs / action (GEMM-side inputs)
  _Float16* obs16 = (_Float16*)alloc((size_t)TT * BB * OBS * sizeof(_Float16));
  _Float16* act16 = (_Float16*)alloc((size_t)TT * BB * ACT * sizeof(_Float16));
  f32_to_f16_kernel<<<4096, 256, 0, stream>>>((const float4*)obs, obs16,
                                              (long)TT * BB * OBS / 4);
  f32_to_f16_kernel<<<256, 256, 0, stream>>>((const float4*)action, act16,
                                             (long)TT * BB * ACT / 4);

  // f16 activation buffers
  _Float16* h    = (_Float16*)alloc((size_t)BB * HID * 2);
  _Float16* s    = (_Float16*)alloc((size_t)BB * STOCH * 2);
  _Float16* xa   = (_Float16*)alloc((size_t)BB * KXA * 2);     // [s, a] padded, tail stays 0
  _Float16* gi   = (_Float16*)alloc((size_t)BB * 3 * HID * 2);
  _Float16* gh   = (_Float16*)alloc((size_t)BB * 3 * HID * 2);
  _Float16* hid1 = (_Float16*)alloc((size_t)BB * DH * 2);
  _Float16* hid2 = (_Float16*)alloc((size_t)BB * DH * 2);
  _Float16* hidc = (_Float16*)alloc((size_t)BB * 1024 * 2);    // fused decoder hidden
  float* pri_out = (float*)alloc((size_t)BB * 2 * STOCH * 4);
  float* pos_out = (float*)alloc((size_t)BB * 2 * STOCH * 4);
  float* accs    = (float*)alloc(2 * 4);   // [0]=sum 0.5*(x-mu)^2 , [1]=sum kl

  hipMemsetAsync(h, 0, (size_t)BB * HID * 2, stream);
  hipMemsetAsync(s, 0, (size_t)BB * STOCH * 2, stream);
  hipMemsetAsync(xa, 0, (size_t)BB * KXA * 2, stream);
  hipMemsetAsync(accs, 0, 2 * 4, stream);

  float* out     = (float*)d_out;
  float* pre_obs = out + 1;
  float* pre_act = pre_obs + (size_t)TT * BB * OBS;
  float* pre_rew = pre_act + (size_t)TT * BB * ACT;

  auto gemm2 = [&](const _Float16* A1, int lda1, int d1, const _Float16* A2, int lda2,
                   const _Float16* Wt, const float* b, void* C, int N, int K, int ldc,
                   int mode, const float* tgt, float* acc) {
    dim3 grid(ceil_div(N, GTN), BB / GTM);
    gemm_wmma_kernel<<<grid, 256, 0, stream>>>(A1, lda1, d1, A2, lda2, Wt, b, C,
                                               N, K, ldc, mode, tgt, acc);
  };
  auto gemm1 = [&](const _Float16* A, int lda, const _Float16* Wt, const float* b, void* C,
                   int N, int K, int ldc, int mode, const float* tgt, float* acc) {
    gemm2(A, lda, K, A, lda, Wt, b, C, N, K, ldc, mode, tgt, acc);
  };

  for (int t = 0; t < TT; ++t) {
    const float*    o_t  = obs    + (size_t)t * BB * OBS;
    const float*    a_t  = action + (size_t)t * BB * ACT;
    const float*    r_t  = reward + (size_t)t * BB;
    const float*    e_t  = eps    + (size_t)t * BB * STOCH;
    const _Float16* o16t = obs16  + (size_t)t * BB * OBS;
    const _Float16* a16t = act16  + (size_t)t * BB * ACT;
    float* po_t = pre_obs + (size_t)t * BB * OBS;
    float* pa_t = pre_act + (size_t)t * BB * ACT;
    float* pr_t = pre_rew + (size_t)t * BB;

    // action_pre on detached previous whole state [h, s] (concat fused into staging)
    gemm2(h, HID, HID, s, STOCH, mlpWt[3][0], pB[3][0], hid1, 512, 1152, 512, 1, nullptr, nullptr);
    gemm1(hid1, 512, mlpWt[3][1], pB[3][1], hid2, 512, 512, 512, 1, nullptr, nullptr);
    gemm1(hid2, 512, mlpWt[3][2], pB[3][2], pa_t, ACT, 512, ACT, 2, a_t, accs);  // + fused NLL

    // GRU: gi = [s,a] @ Wih + bih ; gh = h @ Whh + bhh ; fuse gates
    concat2_kernel<_Float16><<<ceil_div(BB * (STOCH + ACT), 256), 256, 0, stream>>>(
        xa, KXA, s, STOCH, a16t, ACT, BB);
    gemm1(xa, KXA, Wih_t, bih, gi, 3 * HID, KXA, 3 * HID, 0, nullptr, nullptr);
    gemm1(h, HID, Whh_t, bhh, gh, 3 * HID, HID, 3 * HID, 0, nullptr, nullptr);
    gru_update_kernel<<<ceil_div(BB * HID, 256), 256, 0, stream>>>(h, gi, gh);

    // prior(h2) and post([h2, o])
    gemm1(h, HID, mlpWt[1][0], pB[1][0], hid1, 512, 1024, 512, 1, nullptr, nullptr);
    gemm1(hid1, 512, mlpWt[1][1], pB[1][1], hid2, 512, 512, 512, 1, nullptr, nullptr);
    gemm1(hid2, 512, mlpWt[1][2], pB[1][2], pri_out, 2 * STOCH, 512, 2 * STOCH, 3, nullptr, nullptr);
    gemm2(h, HID, HID, o16t, OBS, mlpWt[0][0], pB[0][0], hid1, 512, 2048, 512, 1, nullptr, nullptr);
    gemm1(hid1, 512, mlpWt[0][1], pB[0][1], hid2, 512, 512, 512, 1, nullptr, nullptr);
    gemm1(hid2, 512, mlpWt[0][2], pB[0][2], pos_out, 2 * STOCH, 512, 2 * STOCH, 3, nullptr, nullptr);
    sample_kl_kernel<<<(BB * STOCH) / 256, 256, 0, stream>>>(pri_out, pos_out, e_t, s, accs + 1);

    // decoders on whole = [h2, s2]: fused obs+reward layer-0 (N=1024)
    gemm2(h, HID, HID, s, STOCH, Wdec0, bdec0, hidc, 1024, 1152, 1024, 1, nullptr, nullptr);
    gemm1(hidc,       1024, mlpWt[2][1], pB[2][1], hid2, 512, 512, 512, 1, nullptr, nullptr);
    gemm1(hidc + 512, 1024, mlpWt[4][1], pB[4][1], hid1, 512, 512, 512, 1, nullptr, nullptr);
    gemm1(hid2, 512, mlpWt[2][2], pB[2][2], po_t, OBS, 512, OBS, 2, o_t, accs);
    gemm1(hid1, 512, mlpWt[4][2], pB[4][2], pr_t, 1, 512, 1, 2, r_t, accs);
  }

  final_loss_kernel<<<1, 1, 0, stream>>>(accs, out);
}
